// MergedColumnParallelLinearWithDelta_28973849379100
// MI455X (gfx1250) — matrix operand order repaired and checked by reading, not compiled, not run on hardware
//
#include <hip/hip_runtime.h>

typedef __attribute__((ext_vector_type(16))) _Float16 v16h;
typedef __attribute__((ext_vector_type(2)))  _Float16 v2h;
typedef __attribute__((ext_vector_type(8)))  float    v8f;
typedef __attribute__((ext_vector_type(8)))  unsigned v8u;

#define T_TOK 1024
#define D_IN  2048
#define O_OUT 4096
#define N_ADP 4
#define PFN   8
#define NTOT  (2*O_OUT)

#define MT 64          // token rows per block (4 x 16 sub-tiles per wave)
#define NT 128         // output cols per block (8 waves x 16 cols)
#define KT 32          // K per step = f16 WMMA K-depth
#define LDK (KT + 4)   // LDS row stride in halves (72B -> 8B aligned)

// ---------------------------------------------------------------------------
// Kernel 1: deterministic stable counting sort of tokens by adapter index.
// perm[T_TOK]: sorted token ids; tmask[T_TOK/16]: adapter bitmask per 16-row tile.
// ---------------------------------------------------------------------------
__global__ __launch_bounds__(1024)
void sort_tokens_kernel(const int* __restrict__ indices,
                        int* __restrict__ perm,
                        int* __restrict__ tmask)
{
    __shared__ int waveCnt[32][N_ADP];
    __shared__ int adBase[N_ADP];
    __shared__ int sperm[T_TOK];

    const int t    = threadIdx.x;
    const int lane = t & 31;
    const int w    = t >> 5;
    const int a    = indices[t];
    const unsigned ltm = (1u << lane) - 1u;

    unsigned bal[N_ADP];
    #pragma unroll
    for (int aa = 0; aa < N_ADP; ++aa)
        bal[aa] = __builtin_amdgcn_ballot_w32(a == aa);
    if (lane == 0) {
        #pragma unroll
        for (int aa = 0; aa < N_ADP; ++aa)
            waveCnt[w][aa] = __builtin_popcount(bal[aa]);
    }
    __syncthreads();
    if (t < N_ADP) {                       // totals per adapter
        int s = 0;
        for (int ww = 0; ww < 32; ++ww) s += waveCnt[ww][t];
        adBase[t] = s;
    }
    __syncthreads();
    if (t == 0) {                          // exclusive prefix over adapters
        int run = 0;
        #pragma unroll
        for (int aa = 0; aa < N_ADP; ++aa) { int c = adBase[aa]; adBase[aa] = run; run += c; }
    }
    __syncthreads();
    int off = adBase[a];
    for (int ww = 0; ww < w; ++ww) off += waveCnt[ww][a];      // stable across waves
    off += __builtin_popcount(bal[a] & ltm);                   // stable within wave
    sperm[off] = t;
    __syncthreads();
    perm[t] = sperm[t];
    if (t < T_TOK / 16) {
        int m = 0;
        #pragma unroll
        for (int i = 0; i < 16; ++i)
            m |= 1 << indices[sperm[t * 16 + i]];
        tmask[t] = m;
    }
}

// ---------------------------------------------------------------------------
// Kernel 2: fused base GEMM + GPTQ-4bit adapter deltas on sorted tokens.
// K-permutation note: magic dequant emits fp16 pairs (q_d, q_{d+4}); both A and
// B tiles are stored with p = (k&24) | ((k&3)<<1) | ((k&7)>>2), so the
// contraction is exact with contiguous fragment reads.
// ---------------------------------------------------------------------------
__global__ __launch_bounds__(256)
void fused_gptq_delta_gemm(
    const float* __restrict__ x,
    const float* __restrict__ bw,
    const float* __restrict__ bias,
    const int*   __restrict__ qw0,
    const int*   __restrict__ qw1,
    const int*   __restrict__ qz0,
    const int*   __restrict__ qz1,
    const float* __restrict__ sc0,
    const float* __restrict__ sc1,
    const int*   __restrict__ indices,
    const int*   __restrict__ perm,
    const int*   __restrict__ tmask,
    float*       __restrict__ out)
{
    __shared__ _Float16 Xs[MT][LDK];
    __shared__ _Float16 Ws[NT][LDK];

    const int tid  = threadIdx.x;
    const int lane = tid & 31;
    const int ni   = tid >> 5;          // wave id = 16-col strip (0..7)
    const int l15  = lane & 15;
    const int h    = lane >> 4;

    const int nBase = blockIdx.x * NT;
    const int mBase = blockIdx.y * MT;

    const int slice = (nBase >= O_OUT) ? 1 : 0;
    const int*   qw = slice ? qw1 : qw0;
    const int*   qz = slice ? qz1 : qz0;
    const float* sc = slice ? sc1 : sc0;

    const int ncol = nBase + ni * 16 + l15;
    const int oc   = ncol - slice * O_OUT;

    // sorted-row gather ids for X staging (2 fixed rows per thread)
    const int rowA = perm[mBase + (tid >> 3)];
    const int rowB = perm[mBase + 32 + (tid >> 3)];

    // per-sub-tile adapter of each A row (lane holds one M row) + tile masks
    int myA[4], tmS[4], blockMask = 0;
    #pragma unroll
    for (int st = 0; st < 4; ++st) {
        myA[st] = indices[perm[mBase + st * 16 + l15]];
        int v   = tmask[(mBase >> 4) + st];
        tmS[st] = __builtin_amdgcn_readfirstlane(v);   // force SGPR -> scalar branches
        blockMask |= tmS[st];
    }

    // per-column magic-dequant constants: v = fma(1024+q, sc, -sc*(1024+z))
    v2h sc2[N_ADP], off2[N_ADP];
    #pragma unroll
    for (int a = 0; a < N_ADP; ++a) {
        float s  = sc[(size_t)a * O_OUT + oc];
        int   zp = qz[(size_t)a * (O_OUT / PFN) + (oc >> 3)];
        float z  = (float)((zp >> ((oc & 7) * 4)) & 0xF);
        _Float16 sh = (_Float16)s;
        _Float16 oh = (_Float16)(-s * (1024.0f + z));
        sc2[a]  = (v2h){sh, sh};
        off2[a] = (v2h){oh, oh};
    }

    v8f acc[4] = {};
    const v16h vzero = {};
    const int h8  = h * 8;
    const int h16 = h * 16;
    const int kbx = (tid & 7) << 2;     // float4 column within K tile (X staging)

    for (int k0 = 0; k0 < D_IN; k0 += KT) {
        // ---- stage gathered x rows (64x32) and base_weight (128x32), K-permuted ----
        {
            int pb = (kbx & 24) | ((kbx >> 2) & 1);
            const float4 va = *(const float4*)&x[(size_t)rowA * D_IN + k0 + kbx];
            const float4 vb = *(const float4*)&x[(size_t)rowB * D_IN + k0 + kbx];
            int r1 = tid >> 3, r2 = 32 + (tid >> 3);
            Xs[r1][pb + 0] = (_Float16)va.x;  Xs[r1][pb + 2] = (_Float16)va.y;
            Xs[r1][pb + 4] = (_Float16)va.z;  Xs[r1][pb + 6] = (_Float16)va.w;
            Xs[r2][pb + 0] = (_Float16)vb.x;  Xs[r2][pb + 2] = (_Float16)vb.y;
            Xs[r2][pb + 4] = (_Float16)vb.z;  Xs[r2][pb + 6] = (_Float16)vb.w;
        }
        #pragma unroll
        for (int i = 0; i < (NT * KT) / (256 * 4); ++i) {
            int slot = tid + i * 256;
            int r = slot >> 3, kb = (slot & 7) << 2;
            const float4 v = *(const float4*)&bw[(size_t)(nBase + r) * D_IN + k0 + kb];
            int pb = (kb & 24) | ((kb >> 2) & 1);
            Ws[r][pb + 0] = (_Float16)v.x;  Ws[r][pb + 2] = (_Float16)v.y;
            Ws[r][pb + 4] = (_Float16)v.z;  Ws[r][pb + 6] = (_Float16)v.w;
        }
        if (k0 + KT < D_IN) {   // global_prefetch_b8 for next K tile
            __builtin_prefetch(&x [(size_t)rowA * D_IN + k0 + KT + kbx], 0, 1);
            __builtin_prefetch(&bw[(size_t)(nBase + (tid >> 1)) * D_IN + k0 + KT + ((tid & 1) << 4)], 0, 1);
        }
        __syncthreads();

        // ---- fragments (perm pre-applied -> contiguous LDS reads) ----
        v16h afr[4];
        #pragma unroll
        for (int st = 0; st < 4; ++st) {
            const _Float16* xr = &Xs[st * 16 + l15][0];
            #pragma unroll
            for (int j = 0; j < 8; ++j) {
                afr[st][j]     = xr[h16 + j];
                afr[st][j + 8] = xr[h16 + 8 + j];
            }
        }
        v16h bfrag;
        {
            const _Float16* wr = &Ws[ni * 16 + l15][0];
            #pragma unroll
            for (int j = 0; j < 16; ++j)
                bfrag[j] = wr[h16 + j];
        }

        // ---- base GEMM: 4 sub-tiles share one B fragment ----
        #pragma unroll
        for (int st = 0; st < 4; ++st)
            acc[st] = __builtin_amdgcn_wmma_f32_16x16x32_f16(false, afr[st], false, bfrag,
                                                             (short)0, acc[st], false, false);

        // ---- deltas: dequant + WMMA only for adapters present (scalar branches) ----
        const int kw = (k0 >> 3) + (h << 1);
        #pragma unroll
        for (int a = 0; a < N_ADP; ++a) {
            if (blockMask & (1 << a)) {
                const int2 wv = *(const int2*)(qw + ((size_t)a * O_OUT + oc) * (D_IN / PFN) + kw);
                v8u du;
                #pragma unroll
                for (int d = 0; d < 4; ++d) {
                    unsigned u0 = (((unsigned)wv.x >> (4 * d)) & 0x000F000Fu) | 0x64006400u;
                    unsigned u1 = (((unsigned)wv.y >> (4 * d)) & 0x000F000Fu) | 0x64006400u;
                    v2h r0 = __builtin_bit_cast(v2h, u0) * sc2[a] + off2[a];   // v_pk_fma_f16
                    v2h r1 = __builtin_bit_cast(v2h, u1) * sc2[a] + off2[a];
                    du[d]     = __builtin_bit_cast(unsigned, r0);
                    du[d + 4] = __builtin_bit_cast(unsigned, r1);
                }
                const v16h bd = __builtin_bit_cast(v16h, du);
                #pragma unroll
                for (int st = 0; st < 4; ++st) {
                    if (tmS[st] == (1 << a)) {
                        // sorted fast path: tile is adapter-uniform, no masking
                        acc[st] = __builtin_amdgcn_wmma_f32_16x16x32_f16(false, afr[st], false, bd,
                                                                         (short)0, acc[st], false, false);
                    } else if (tmS[st] & (1 << a)) {
                        // boundary tile: per-lane dispatch
                        v16h am = (myA[st] == a) ? afr[st] : vzero;
                        acc[st] = __builtin_amdgcn_wmma_f32_16x16x32_f16(false, am, false, bd,
                                                                         (short)0, acc[st], false, false);
                    }
                }
            }
        }
        __syncthreads();
    }

    // ---- scatter store through perm: lane = N col, VGPR v = M row ----
    const int cg = nBase + ni * 16 + l15;
    const float bb = bias[cg];
    #pragma unroll
    for (int st = 0; st < 4; ++st) {
        #pragma unroll
        for (int v = 0; v < 8; ++v) {
            int rs = mBase + st * 16 + h * 8 + v;    // sorted row
            int rg = perm[rs];                       // original token row
            out[(size_t)rg * NTOT + cg] = acc[st][v] + bb;
        }
    }
}

extern "C" void kernel_launch(void* const* d_in, const int* in_sizes, int n_in,
                              void* d_out, int out_size, void* d_ws, size_t ws_size,
                              hipStream_t stream) {
    const float* x    = (const float*)d_in[0];
    const float* bw   = (const float*)d_in[1];
    const float* bias = (const float*)d_in[2];
    const int*   qw0  = (const int*)  d_in[3];
    const int*   qw1  = (const int*)  d_in[4];
    const int*   qz0  = (const int*)  d_in[5];
    const int*   qz1  = (const int*)  d_in[6];
    const float* sc0  = (const float*)d_in[7];
    const float* sc1  = (const float*)d_in[8];
    // d_in[9], d_in[10] = g_idx (all zeros -> single quant group, unused)
    const int*   idx  = (const int*)  d_in[11];
    float*       out  = (float*)      d_out;

    int* perm  = (int*)d_ws;                 // T_TOK ints
    int* tmask = (int*)d_ws + T_TOK;         // T_TOK/16 ints  (needs 4352 B of ws)

    sort_tokens_kernel<<<1, 1024, 0, stream>>>(idx, perm, tmask);

    dim3 grid(NTOT / NT, T_TOK / MT);        // 64 x 16 tiles
    dim3 block(256);
    fused_gptq_delta_gemm<<<grid, block, 0, stream>>>(
        x, bw, bias, qw0, qw1, qz0, qz1, sc0, sc1, idx, perm, tmask, out);
}
